// BiLstmCrf_29033978921244
// MI455X (gfx1250) — matrix operand order repaired and checked by reading, not compiled, not run on hardware
//
#include <hip/hip_runtime.h>

typedef __attribute__((ext_vector_type(16))) __bf16 v16bf;
typedef __attribute__((ext_vector_type(8)))  float  v8f;

#define BDIM 64
#define TDIM 256
#define EDIM 512
#define HDIM 256
#define NTAGS 6
#define TAG_START 4
#define TAG_STOP 5
#define NEGV (-1000000.0f)

// ---------- helpers ----------
__device__ __forceinline__ unsigned short f2bf(float f) {
  unsigned int u = __float_as_uint(f);
  unsigned int r = u + 0x7FFFu + ((u >> 16) & 1u);
  return (unsigned short)(r >> 16);
}

__device__ __forceinline__ float sigm(float x) { return 1.0f / (1.0f + expf(-x)); }

__device__ __forceinline__ v8f zero8() {
  v8f z = {0.f, 0.f, 0.f, 0.f, 0.f, 0.f, 0.f, 0.f};
  return z;
}

__device__ __forceinline__ v8f wmma_bf16(v16bf a, v16bf b, v8f c) {
  return __builtin_amdgcn_wmma_f32_16x16x32_bf16(false, a, false, b, (short)0, c,
                                                 false, false);
}

// A tile: 16 (M) x 32 (K) bf16 from row-major X[ld], contiguous k layout (k = 16h + s):
// lanes 0-15 row r, K 0..15; lanes 16-31 row r, K 16..31
__device__ __forceinline__ v16bf load_a_tile(const unsigned short* X, int ld,
                                             int rowBase, int kbase, int lane) {
  int hf = lane >> 4, r = lane & 15;
  union { v16bf v; unsigned int u[8]; } A;
  const unsigned int* p = (const unsigned int*)(X + (long)(rowBase + r) * ld + kbase);
  int ko = hf * 8; // uint units (16 bf16)
#pragma unroll
  for (int i = 0; i < 4; ++i) {
    A.u[i]     = p[ko + i];
    A.u[4 + i] = p[4 + ko + i];
  }
  return A.v;
}

// B tile: 32 (K) x 16 (N) bf16 where B[k][n] = W[n][k], W row-major [N,K]:
// lane n holds column n, lanes 0-15 K=0..15, lanes 16-31 K=16..31
__device__ __forceinline__ v16bf load_b_tile(const unsigned short* W, int ldw,
                                             int colBase, int kbase, int lane) {
  int khalf = lane >> 4, n = lane & 15;
  return *(const v16bf*)(W + (long)(colBase + n) * ldw + kbase + khalf * 16);
}

// ---------- kernels ----------

// x[t][b][:] = bf16(embed[sentences[b][t]])
__global__ __launch_bounds__(256) void gather_kernel(const int* __restrict__ sentences,
                                                     const float* __restrict__ embed,
                                                     unsigned short* __restrict__ xbf) {
  int row = blockIdx.x;            // t*64 + b
  int b = row & 63, t = row >> 6;
  int v = sentences[b * TDIM + t];
  const float* src = embed + (long)v * EDIM;
  for (int e = threadIdx.x; e < EDIM; e += 256)
    xbf[(long)row * EDIM + e] = f2bf(src[e]);
}

__global__ __launch_bounds__(256) void f32_to_bf16_kernel(const float* __restrict__ src,
                                                          unsigned short* __restrict__ dst,
                                                          long n) {
  long i = (long)blockIdx.x * 256 + threadIdx.x;
  long stride = (long)gridDim.x * 256;
  for (; i < n; i += stride) dst[i] = f2bf(src[i]);
}

__global__ __launch_bounds__(256) void biascomb_kernel(const float* __restrict__ bih,
                                                       const float* __restrict__ bhh,
                                                       float* __restrict__ out) {
  int i = blockIdx.x * 256 + threadIdx.x;
  if (i < 3 * 2 * 1024) out[i] = bih[i] + bhh[i];
}

__global__ __launch_bounds__(256) void zero_state_kernel(float* __restrict__ c,
                                                         unsigned short* __restrict__ h) {
  int i = blockIdx.x * 256 + threadIdx.x; // n = 2*64*256 = 32768
  c[i] = 0.f;
  h[i] = 0;
}

// out[z][M,N] = Xbf[M,K] @ Wbf[z][N,K]^T + bias[z][N]
// wave tile 32x64 (8 WMMA per 2 A + 4 B tile loads), software-pipelined K loop
__global__ __launch_bounds__(256) void gemm_xw_kernel(const unsigned short* __restrict__ X,
                                                      const unsigned short* __restrict__ W,
                                                      const float* __restrict__ bias,
                                                      float* __restrict__ out,
                                                      int N, int K,
                                                      long wStrideZ, long bStrideZ, long oStrideZ) {
  int z = blockIdx.z;
  W += (long)z * wStrideZ;
  bias += (long)z * bStrideZ;
  out += (long)z * oStrideZ;

  int lane = threadIdx.x & 31;
  int wave = threadIdx.x >> 5;        // 0..7
  int wm = wave >> 2, wn = wave & 3;  // 2 x 4 wave grid
  int rowBase = blockIdx.x * 64 + wm * 32;
  int colBase = blockIdx.y * 256 + wn * 64;

  v8f acc[2][4];
#pragma unroll
  for (int mi = 0; mi < 2; ++mi)
#pragma unroll
    for (int j = 0; j < 4; ++j) acc[mi][j] = zero8();

  // pipeline primers
  v16bf a0 = load_a_tile(X, K, rowBase, 0, lane);
  v16bf a1 = load_a_tile(X, K, rowBase + 16, 0, lane);
  v16bf bc = load_b_tile(W, K, colBase, 0, lane);

  for (int kt = 0; kt < K; kt += 32) {
    int ktn = (kt + 32 < K) ? kt + 32 : 0; // wrap: last prefetch is discarded
    v16bf a0n = load_a_tile(X, K, rowBase, ktn, lane);
    v16bf a1n = load_a_tile(X, K, rowBase + 16, ktn, lane);
#pragma unroll
    for (int j = 0; j < 4; ++j) {
      int jn = j + 1, kb = kt;
      if (jn == 4) { jn = 0; kb = ktn; }
      v16bf bn = load_b_tile(W, K, colBase + jn * 16, kb, lane);
      acc[0][j] = wmma_bf16(a0, bc, acc[0][j]);
      acc[1][j] = wmma_bf16(a1, bc, acc[1][j]);
      bc = bn;
    }
    a0 = a0n;
    a1 = a1n;
  }

  int hf = lane >> 4, n15 = lane & 15;
#pragma unroll
  for (int mi = 0; mi < 2; ++mi) {
#pragma unroll
    for (int j = 0; j < 4; ++j) {
      int col = colBase + j * 16 + n15;
      float bv = bias[col];
#pragma unroll
      for (int v = 0; v < 8; ++v) {
        int row = rowBase + mi * 16 + v + hf * 8;
        out[(long)row * N + col] = acc[mi][j][v] + bv;
      }
    }
  }
}

// one timestep, both directions (grid.y), fused h@Whh^T + LSTM cell
__global__ __launch_bounds__(256) void lstm_step_kernel(const unsigned short* __restrict__ whh_bf,
                                                        const float* __restrict__ gates_x,
                                                        unsigned short* __restrict__ hbf,
                                                        float* __restrict__ cstate,
                                                        float* __restrict__ xout,
                                                        int layer, int step) {
  __shared__ float lds[16 * 1024]; // 64 KB gate tile

  int dir = blockIdx.y;
  int t = dir ? (TDIM - 1 - step) : step;
  const unsigned short* W = whh_bf + ((long)layer * 2 + dir) * (4 * HDIM * HDIM);
  const float* gx = gates_x + (long)dir * ((long)TDIM * BDIM * 4 * HDIM)
                            + (long)t * (BDIM * 4 * HDIM);
  unsigned short* hs = hbf + dir * (BDIM * HDIM);
  float* cs = cstate + dir * (BDIM * HDIM);

  int rowBase = blockIdx.x * 16; // batch rows
  int lane = threadIdx.x & 31;
  int wave = threadIdx.x >> 5;   // col slab wave*128

  // warm L2/WGP$ for the gate rows consumed after the GEMM
  __builtin_prefetch(&gx[(long)(rowBase + (threadIdx.x & 15)) * 1024 +
                         (threadIdx.x >> 4) * 64], 0, 1);

  v8f acc[8];
#pragma unroll
  for (int j = 0; j < 8; ++j) acc[j] = zero8();

  // pipeline primers
  v16bf a = load_a_tile(hs, HDIM, rowBase, 0, lane);
  v16bf bc = load_b_tile(W, HDIM, wave * 128, 0, lane);

  for (int kt = 0; kt < HDIM; kt += 32) {
    int ktn = (kt + 32 < HDIM) ? kt + 32 : 0;
    v16bf an = load_a_tile(hs, HDIM, rowBase, ktn, lane);
#pragma unroll
    for (int j = 0; j < 8; ++j) {
      int jn = j + 1, kb = kt;
      if (jn == 8) { jn = 0; kb = ktn; }
      v16bf bn = load_b_tile(W, HDIM, wave * 128 + jn * 16, kb, lane);
      acc[j] = wmma_bf16(a, bc, acc[j]);
      bc = bn;
    }
    a = an;
  }

  int hf = lane >> 4, n15 = lane & 15;
#pragma unroll
  for (int j = 0; j < 8; ++j) {
    int col = wave * 128 + j * 16 + n15;
#pragma unroll
    for (int v = 0; v < 8; ++v) {
      int r = v + hf * 8;
      lds[r * 1024 + col] = acc[j][v] + gx[(long)(rowBase + r) * 1024 + col];
    }
  }
  __syncthreads();

  // cell: 16 rows x 256 h-values = 4096 elements / 256 threads
#pragma unroll
  for (int i = 0; i < 16; ++i) {
    int e = threadIdx.x + 256 * i;
    int r = e >> 8, j = e & 255;
    float ig = lds[r * 1024 + j];
    float fg = lds[r * 1024 + 256 + j];
    float gg = lds[r * 1024 + 512 + j];
    float og = lds[r * 1024 + 768 + j];
    int b = rowBase + r;
    float c = sigm(fg) * cs[b * HDIM + j] + sigm(ig) * tanhf(gg);
    float h = sigm(og) * tanhf(c);
    cs[b * HDIM + j] = c;
    hs[b * HDIM + j] = f2bf(h);
    xout[((long)t * BDIM + b) * EDIM + dir * HDIM + j] = h;
  }
}

// energy[b][t] = sum_k tanh(proj[row,k]) * aw2[k]   (wave per row; ab2 cancels in softmax)
__global__ __launch_bounds__(256) void energy_kernel(const float* __restrict__ proj,
                                                     const float* __restrict__ aw2,
                                                     float* __restrict__ energy) {
  int wave = threadIdx.x >> 5, lane = threadIdx.x & 31;
  int row = blockIdx.x * 8 + wave; // t*64 + b
  float s = 0.f;
  for (int k = lane; k < EDIM; k += 32)
    s += tanhf(proj[(long)row * EDIM + k]) * aw2[k];
#pragma unroll
  for (int o = 16; o; o >>= 1) s += __shfl_xor(s, o, 32);
  if (lane == 0) {
    int b = row & 63, t = row >> 6;
    energy[b * TDIM + t] = s;
  }
}

__global__ __launch_bounds__(256) void softmax_kernel(const float* __restrict__ energy,
                                                      float* __restrict__ wts) {
  __shared__ float red[256];
  int b = blockIdx.x, t = threadIdx.x;
  float e = energy[b * TDIM + t];
  red[t] = e;
  __syncthreads();
  for (int o = 128; o; o >>= 1) {
    if (t < o) red[t] = fmaxf(red[t], red[t + o]);
    __syncthreads();
  }
  float m = red[0];
  __syncthreads();
  float ex = expf(e - m);
  red[t] = ex;
  __syncthreads();
  for (int o = 128; o; o >>= 1) {
    if (t < o) red[t] += red[t + o];
    __syncthreads();
  }
  wts[b * TDIM + t] = ex / red[0];
}

// feats[b][t][tag] = hb[tag] + (1+w[b][t]) * sum_k lstm_out[row,k]*hw[tag,k]
__global__ __launch_bounds__(256) void feats_kernel(const float* __restrict__ lstm_out,
                                                    const float* __restrict__ wts,
                                                    const float* __restrict__ hw,
                                                    const float* __restrict__ hb,
                                                    float* __restrict__ feats) {
  int wave = threadIdx.x >> 5, lane = threadIdx.x & 31;
  int row = blockIdx.x * 8 + wave;
  int b = row & 63, t = row >> 6;
  float scale = 1.0f + wts[b * TDIM + t];
  float s[NTAGS] = {0.f, 0.f, 0.f, 0.f, 0.f, 0.f};
  for (int k = lane; k < EDIM; k += 32) {
    float v = lstm_out[(long)row * EDIM + k] * scale;
#pragma unroll
    for (int tg = 0; tg < NTAGS; ++tg) s[tg] += v * hw[tg * EDIM + k];
  }
#pragma unroll
  for (int tg = 0; tg < NTAGS; ++tg) {
    float x = s[tg];
#pragma unroll
    for (int o = 16; o; o >>= 1) x += __shfl_xor(x, o, 32);
    if (lane == 0) feats[((long)b * TDIM + t) * NTAGS + tg] = x + hb[tg];
  }
}

// CRF forward + gold + mean loss; thread per batch element
__global__ __launch_bounds__(64) void crf_kernel(const float* __restrict__ feats,
                                                 const float* __restrict__ trans,
                                                 const int* __restrict__ tags,
                                                 const float* __restrict__ mask,
                                                 float* __restrict__ out) {
  __shared__ float red[64];
  int b = threadIdx.x;
  float fv[NTAGS];
#pragma unroll
  for (int i = 0; i < NTAGS; ++i) fv[i] = NEGV;
  fv[TAG_START] = 0.f;

  float gold = 0.f, msum = 0.f;
  int prev = TAG_START;

  for (int t = 0; t < TDIM; ++t) {
    float feat[NTAGS];
#pragma unroll
    for (int i = 0; i < NTAGS; ++i) feat[i] = feats[((long)b * TDIM + t) * NTAGS + i];
    float m = mask[b * TDIM + t];
    float nxt[NTAGS];
#pragma unroll
    for (int i = 0; i < NTAGS; ++i) {
      float mx = -3.4e38f, vals[NTAGS];
#pragma unroll
      for (int j = 0; j < NTAGS; ++j) {
        vals[j] = fv[j] + trans[i * NTAGS + j] + feat[i];
        mx = fmaxf(mx, vals[j]);
      }
      float s = 0.f;
#pragma unroll
      for (int j = 0; j < NTAGS; ++j) s += expf(vals[j] - mx);
      nxt[i] = mx + logf(s);
    }
#pragma unroll
    for (int i = 0; i < NTAGS; ++i) fv[i] = (m > 0.f) ? nxt[i] : fv[i];

    int cur = tags[b * TDIM + t];
    gold += (feat[cur] + trans[cur * NTAGS + prev]) * m;
    prev = cur;
    msum += m;
  }

  // forward score: logsumexp(fv + trans[STOP])
  float mx = -3.4e38f, vals[NTAGS];
#pragma unroll
  for (int j = 0; j < NTAGS; ++j) {
    vals[j] = fv[j] + trans[TAG_STOP * NTAGS + j];
    mx = fmaxf(mx, vals[j]);
  }
  float s = 0.f;
#pragma unroll
  for (int j = 0; j < NTAGS; ++j) s += expf(vals[j] - mx);
  float fscore = mx + logf(s);

  int sl = (int)(msum + 0.5f);
  int last = (sl == 0) ? TAG_START : tags[b * TDIM + (sl - 1)];
  gold += trans[TAG_STOP * NTAGS + last];

  red[b] = fscore - gold;
  __syncthreads();
  for (int o = 32; o; o >>= 1) {
    if (b < o) red[b] += red[b + o];
    __syncthreads();
  }
  if (b == 0) out[0] = red[0] / (float)BDIM;
}

// ---------- host ----------
extern "C" void kernel_launch(void* const* d_in, const int* in_sizes, int n_in,
                              void* d_out, int out_size, void* d_ws, size_t ws_size,
                              hipStream_t stream) {
  (void)in_sizes; (void)n_in; (void)out_size; (void)ws_size;
  const int*   sentences = (const int*)d_in[0];
  const int*   tags      = (const int*)d_in[1];
  const float* mask      = (const float*)d_in[2];
  const float* embed     = (const float*)d_in[3];
  const float* w_ih      = (const float*)d_in[4];
  const float* w_hh      = (const float*)d_in[5];
  const float* b_ih      = (const float*)d_in[6];
  const float* b_hh      = (const float*)d_in[7];
  const float* aw1       = (const float*)d_in[8];
  const float* ab1       = (const float*)d_in[9];
  const float* aw2       = (const float*)d_in[10];
  /* ab2 (d_in[11]) cancels in softmax */
  const float* hw        = (const float*)d_in[12];
  const float* hb        = (const float*)d_in[13];
  const float* trans     = (const float*)d_in[14];
  float* out = (float*)d_out;

  char* p = (char*)d_ws;
  auto alloc = [&](size_t bytes) {
    char* r = p;
    p += (bytes + 255) & ~((size_t)255);
    return r;
  };
  const long MROWS = (long)TDIM * BDIM; // 16384
  unsigned short* xbf    = (unsigned short*)alloc(MROWS * EDIM * 2);
  float*          gates  = (float*)alloc(2ull * MROWS * 1024 * 4);
  float*          outA   = (float*)alloc(MROWS * EDIM * 4);
  float*          outB   = (float*)alloc(MROWS * EDIM * 4);
  unsigned short* wih_bf = (unsigned short*)alloc(3ull * 2 * 1024 * EDIM * 2);
  unsigned short* whh_bf = (unsigned short*)alloc(3ull * 2 * 1024 * HDIM * 2);
  unsigned short* aw1_bf = (unsigned short*)alloc((size_t)EDIM * EDIM * 2);
  float*          biasc  = (float*)alloc(3ull * 2 * 1024 * 4);
  float*          cstate = (float*)alloc(2ull * BDIM * HDIM * 4);
  unsigned short* hbf    = (unsigned short*)alloc(2ull * BDIM * HDIM * 2);
  float*          energy = (float*)alloc(MROWS * 4);
  float*          wts    = (float*)alloc(MROWS * 4);
  float*          featsb = (float*)alloc(MROWS * NTAGS * 4);
  float*          proj   = gates; // reused after LSTM layers

  // weight / bias prep
  f32_to_bf16_kernel<<<2048, 256, 0, stream>>>(w_ih, wih_bf, 3L * 2 * 1024 * EDIM);
  f32_to_bf16_kernel<<<2048, 256, 0, stream>>>(w_hh, whh_bf, 3L * 2 * 1024 * HDIM);
  f32_to_bf16_kernel<<<1024, 256, 0, stream>>>(aw1, aw1_bf, (long)EDIM * EDIM);
  biascomb_kernel<<<24, 256, 0, stream>>>(b_ih, b_hh, biasc);

  // embedding -> bf16 [T,B,E]
  gather_kernel<<<(int)MROWS, 256, 0, stream>>>(sentences, embed, xbf);

  for (int layer = 0; layer < 3; ++layer) {
    // gates_x[dir] = Xbf @ Wih[layer][dir]^T + bias (both dirs in one launch)
    {
      dim3 grid(MROWS / 64, 1024 / 256, 2);
      gemm_xw_kernel<<<grid, 256, 0, stream>>>(
          xbf, wih_bf + (long)layer * 2 * 1024 * EDIM, biasc + layer * 2048, gates,
          1024, EDIM,
          (long)1024 * EDIM, 1024L, MROWS * 1024L);
    }
    zero_state_kernel<<<128, 256, 0, stream>>>(cstate, hbf);
    float* xout = (layer == 1) ? outB : outA;
    for (int step = 0; step < TDIM; ++step) {
      dim3 grid(BDIM / 16, 2);
      lstm_step_kernel<<<grid, 256, 0, stream>>>(whh_bf, gates, hbf, cstate, xout,
                                                 layer, step);
    }
    // layer output -> bf16 input for next layer / attention
    f32_to_bf16_kernel<<<4096, 256, 0, stream>>>(xout, xbf, MROWS * EDIM);
  }
  float* lstm_out = outA; // layer 2 output

  // attention projection: proj = lstm_out_bf16 @ aw1^T + ab1
  {
    dim3 grid(MROWS / 64, EDIM / 256, 1);
    gemm_xw_kernel<<<grid, 256, 0, stream>>>(xbf, aw1_bf, ab1, proj,
                                             EDIM, EDIM, 0L, 0L, 0L);
  }
  energy_kernel<<<(int)(MROWS / 8), 256, 0, stream>>>(proj, aw2, energy);
  softmax_kernel<<<BDIM, 256, 0, stream>>>(energy, wts);
  feats_kernel<<<(int)(MROWS / 8), 256, 0, stream>>>(lstm_out, wts, hw, hb, featsb);
  crf_kernel<<<1, 64, 0, stream>>>(featsb, trans, tags, mask, out);
}
